// Layers_LSTM_88536455840382
// MI455X (gfx1250) — matrix operand order repaired and checked by reading, not compiled, not run on hardware
//
#include <hip/hip_runtime.h>
#include <hip/hip_bf16.h>

// ---------------------------------------------------------------------------
// 2-layer LSTM for MI455X (gfx1250, wave32, WMMA).
//   L=512, B=64, C=H=1024, 4H=4096.
// Persistent cooperative kernel per layer: 64 blocks x 256 threads.
// Block wg owns 16 hidden columns -> 64 gate rows; its W_ih / W_hh slices are
// held in LDS (bf16, ~264KB) for the whole sequence (CDNA5: 320KB per WGP).
// Gates computed with v_wmma_f32_16x16x32_bf16; cell state lives in registers.
// ---------------------------------------------------------------------------

#define L_SEQ  512
#define BATCH  64
#define HDIM   1024
#define NBLK   64
#define NTHR   256
#define WPITCH 1032   // 1024 + 8 bf16 pad (bank-conflict mitigation)
#define GPITCH 17     // padded gate-tile row pitch (floats)

typedef __attribute__((ext_vector_type(16))) __bf16 v16bf;
typedef __attribute__((ext_vector_type(8)))  __bf16 v8bf;
typedef __attribute__((ext_vector_type(8)))  float  v8f;

union V16U { v16bf v; v8bf h[2]; };

// A-matrix (16x32 bf16) fragment from row-major global memory.
// ISA layout: lanes 0-15 hold row (lane), K = {0..7, 16..23};
//             lanes 16-31 hold row (lane-16), K = {8..15, 24..31}.
__device__ __forceinline__ v16bf load_a_frag(const __bf16* __restrict__ base,
                                             int row0, int pitch, int k0, int lane) {
  const int r  = row0 + (lane & 15);
  const int kb = (lane >> 4) << 3;               // 0 or 8
  const __bf16* p = base + r * pitch + k0 + kb;
  V16U u;
  u.h[0] = *(const v8bf*)(p);                    // K = k0+kb .. +7
  u.h[1] = *(const v8bf*)(p + 16);               // K = k0+16+kb .. +7
  return u.v;
}

// B-matrix (32x16 bf16) fragment from LDS, weights stored N-major:
// lds[srow * WPITCH + k] = W[gate_row srow][k]. Lane l: column n = l&15,
// K = (l<16) ? {k0..k0+7, k0+16..k0+23} : {k0+8..k0+15, k0+24..k0+31}.
__device__ __forceinline__ v16bf load_b_frag(const __bf16* __restrict__ ldsw,
                                             int srow0, int k0, int lane) {
  const int n  = lane & 15;
  const int kb = (lane >> 4) << 3;               // 0 or 8
  const __bf16* p = ldsw + (srow0 + n) * WPITCH + k0 + kb;
  V16U u;
  u.h[0] = *(const v8bf*)(p);
  u.h[1] = *(const v8bf*)(p + 16);
  return u.v;
}

__device__ __forceinline__ v8f wmma_bf16(v16bf a, v16bf b, v8f c) {
  // (neg_a, A, neg_b, B, c_mod, C, reuse_a, reuse_b)
  return __builtin_amdgcn_wmma_f32_16x16x32_bf16(false, a, false, b,
                                                 (short)0, c, false, false);
}

// Monotonic-counter grid barrier (all NBLK blocks co-resident).
__device__ __forceinline__ void grid_barrier(unsigned* cnt, unsigned target) {
  __threadfence();                // release this thread's global stores
  __syncthreads();
  if (threadIdx.x == 0) {
    atomicAdd(cnt, 1u);
    while (atomicAdd(cnt, 0u) < target) { __builtin_amdgcn_s_sleep(2); }
  }
  __syncthreads();
  __threadfence();                // acquire: invalidate stale WGP$ lines
}

__device__ __forceinline__ float sigmoidf_dev(float x) {
  return 1.0f / (1.0f + __expf(-x));
}

__global__ void __launch_bounds__(NTHR, 1)
lstm_layer_kernel(const __bf16* __restrict__ xin,   // [L, B, 1024] bf16
                  const float*  __restrict__ W_ih,  // [4096, 1024] f32
                  const float*  __restrict__ W_hh,  // [4096, 1024] f32
                  const float*  __restrict__ bias,  // [4096] f32
                  __bf16* __restrict__ hbuf0,       // [B, H] bf16 (t even in)
                  __bf16* __restrict__ hbuf1,       // [B, H] bf16
                  __bf16* __restrict__ yout_bf,     // [L, B, H] bf16 or null
                  float*  __restrict__ yout_f32,    // [L, B, H] f32 or null
                  float*  __restrict__ hn_out,      // [B*H] f32
                  float*  __restrict__ cn_out,      // [B*H] f32
                  unsigned* __restrict__ cnt)
{
  extern __shared__ char smem[];
  __bf16* lds_wih  = (__bf16*)smem;                         // 64 * WPITCH bf16
  __bf16* lds_whh  = lds_wih + 64 * WPITCH;                 // 64 * WPITCH bf16
  float*  lds_gate = (float*)(lds_whh + 64 * WPITCH);       // 4 * 64 * GPITCH f32
  float*  lds_bias = lds_gate + 4 * 64 * GPITCH;            // 64 f32

  const int tid  = threadIdx.x;
  const int lane = tid & 31;
  const int wave = tid >> 5;          // 0..7 (wave32)
  const int wg   = blockIdx.x;        // owns H columns [wg*16, wg*16+16)
  const int col0 = wg << 4;

  // ---- stage weight slices into LDS (f32 -> bf16), rows = q*16 + j ----
  for (int idx = tid; idx < 64 * 1024; idx += NTHR) {
    const int sr = idx >> 10;                         // slice row 0..63
    const int k  = idx & 1023;
    const int g  = ((sr >> 4) << 10) + col0 + (sr & 15);   // q*1024 + col0 + j
    lds_wih[sr * WPITCH + k] = (__bf16)W_ih[g * 1024 + k];
    lds_whh[sr * WPITCH + k] = (__bf16)W_hh[g * 1024 + k];
  }
  if (tid < 64) {
    const int g = ((tid >> 4) << 10) + col0 + (tid & 15);
    lds_bias[tid] = bias[g];
  }

  // ---- each thread owns 4 cells: batch b_own, columns j_own0..+3 ----
  const int b_own  = tid >> 2;
  const int j_own0 = (tid & 3) << 2;
  for (int u = 0; u < 4; ++u)
    hbuf0[b_own * HDIM + col0 + j_own0 + u] = (__bf16)0.0f;   // h(-1) = 0

  float c_state[4] = {0.f, 0.f, 0.f, 0.f};
  float h_last[4]  = {0.f, 0.f, 0.f, 0.f};

  const int gi    = wave >> 1;         // gate index: 0=i 1=f 2=g 3=o
  const int m0    = (wave & 1) << 5;   // batch-tile base: 0 or 32
  const int srow0 = gi << 4;

  unsigned phase = 1;
  grid_barrier(cnt, NBLK * phase);     // init (weights + h0) visible

  for (int t = 0; t < L_SEQ; ++t) {
    const __bf16* xt   = xin + (size_t)t * (BATCH * HDIM);
    const __bf16* h_in = (t & 1) ? hbuf1 : hbuf0;
    __bf16*       h_out= (t & 1) ? hbuf0 : hbuf1;

    if (t + 1 < L_SEQ)   // warm caches for next step's activations
      __builtin_prefetch(xin + (size_t)(t + 1) * (BATCH * HDIM) + tid * 256, 0, 0);

    v8f acc0 = {0.f,0.f,0.f,0.f,0.f,0.f,0.f,0.f};
    v8f acc1 = acc0;

    // gates = x_t @ W_ih^T  (K = 1024)
    #pragma unroll 2
    for (int k0 = 0; k0 < 1024; k0 += 32) {
      v16bf bf = load_b_frag(lds_wih, srow0, k0, lane);
      v16bf a0 = load_a_frag(xt, m0,      HDIM, k0, lane);
      v16bf a1 = load_a_frag(xt, m0 + 16, HDIM, k0, lane);
      acc0 = wmma_bf16(a0, bf, acc0);
      acc1 = wmma_bf16(a1, bf, acc1);
    }
    //      += h_{t-1} @ W_hh^T
    #pragma unroll 2
    for (int k0 = 0; k0 < 1024; k0 += 32) {
      v16bf bf = load_b_frag(lds_whh, srow0, k0, lane);
      v16bf a0 = load_a_frag(h_in, m0,      HDIM, k0, lane);
      v16bf a1 = load_a_frag(h_in, m0 + 16, HDIM, k0, lane);
      acc0 = wmma_bf16(a0, bf, acc0);
      acc1 = wmma_bf16(a1, bf, acc1);
    }

    // ---- stage gate tiles (C-layout: VGPR v -> M = v or v+8) ----
    {
      const int nn = lane & 15;
      const int rb = (lane < 16) ? 0 : 8;
      float* gb = lds_gate + gi * (64 * GPITCH);
      #pragma unroll
      for (int v = 0; v < 8; ++v) {
        gb[(m0 + rb + v)      * GPITCH + nn] = acc0[v];
        gb[(m0 + 16 + rb + v) * GPITCH + nn] = acc1[v];
      }
    }
    __syncthreads();

    // ---- elementwise LSTM cell (faithful: h uses PREVIOUS c) ----
    #pragma unroll
    for (int u = 0; u < 4; ++u) {
      const int j = j_own0 + u;
      const int gr = b_own * GPITCH + j;
      const float vi = lds_gate[0 * (64 * GPITCH) + gr] + lds_bias[j];
      const float vf = lds_gate[1 * (64 * GPITCH) + gr] + lds_bias[16 + j];
      const float vg = lds_gate[2 * (64 * GPITCH) + gr] + lds_bias[32 + j];
      const float vo = lds_gate[3 * (64 * GPITCH) + gr] + lds_bias[48 + j];
      const float ig = sigmoidf_dev(vi);
      const float fg = sigmoidf_dev(vf);
      const float gg = tanhf(vg);
      const float og = sigmoidf_dev(vo);
      const float hn = og * tanhf(c_state[u]);          // previous cell state
      const float cn = fg * c_state[u] + ig * gg;
      c_state[u] = cn;
      h_last[u]  = hn;
      const int gcol = b_own * HDIM + col0 + j;
      h_out[gcol] = (__bf16)hn;
      if (yout_bf)  yout_bf [(size_t)t * (BATCH * HDIM) + gcol] = (__bf16)hn;
      if (yout_f32) yout_f32[(size_t)t * (BATCH * HDIM) + gcol] = hn;
    }

    ++phase;
    grid_barrier(cnt, NBLK * phase);   // h_out visible device-wide
  }

  // final hidden / cell state
  #pragma unroll
  for (int u = 0; u < 4; ++u) {
    const int gcol = b_own * HDIM + col0 + j_own0 + u;
    hn_out[gcol] = h_last[u];
    cn_out[gcol] = c_state[u];
  }
}

// f32 -> bf16 activation convert + barrier-counter reset (runs first).
__global__ void convert_init_kernel(const float* __restrict__ x,
                                    __bf16* __restrict__ x_bf, size_t n,
                                    unsigned* __restrict__ counters) {
  size_t i = (size_t)blockIdx.x * blockDim.x + threadIdx.x;
  const size_t stride = (size_t)gridDim.x * blockDim.x;
  for (; i < n; i += stride) x_bf[i] = (__bf16)x[i];
  if (blockIdx.x == 0 && threadIdx.x < 2) counters[threadIdx.x] = 0u;
}

extern "C" void kernel_launch(void* const* d_in, const int* in_sizes, int n_in,
                              void* d_out, int out_size, void* d_ws, size_t ws_size,
                              hipStream_t stream) {
  const float* x     = (const float*)d_in[0];
  const float* W_ih0 = (const float*)d_in[1];
  const float* W_hh0 = (const float*)d_in[2];
  const float* b0    = (const float*)d_in[3];
  const float* W_ih1 = (const float*)d_in[4];
  const float* W_hh1 = (const float*)d_in[5];
  const float* b1    = (const float*)d_in[6];

  const size_t seq_elems = (size_t)L_SEQ * BATCH * HDIM;   // 33,554,432
  const size_t hc_elems  = (size_t)BATCH * HDIM;           // 65,536

  float* out = (float*)d_out;
  float* y1  = out;                         // [L, B, H]
  float* h_n = out + seq_elems;             // [2, B, H]
  float* c_n = h_n + 2 * hc_elems;          // [2, B, H]

  char* ws = (char*)d_ws;
  __bf16* x_bf  = (__bf16*)ws;  ws += seq_elems * sizeof(__bf16);
  __bf16* y0_bf = (__bf16*)ws;  ws += seq_elems * sizeof(__bf16);
  __bf16* hb0   = (__bf16*)ws;  ws += hc_elems * sizeof(__bf16);
  __bf16* hb1   = (__bf16*)ws;  ws += hc_elems * sizeof(__bf16);
  unsigned* counters = (unsigned*)ws;

  convert_init_kernel<<<512, 256, 0, stream>>>(x, x_bf, seq_elems, counters);

  const size_t shmem = (size_t)(2 * 64 * WPITCH) * sizeof(__bf16)
                     + (size_t)(4 * 64 * GPITCH + 64) * sizeof(float);

  lstm_layer_kernel<<<NBLK, NTHR, shmem, stream>>>(
      x_bf, W_ih0, W_hh0, b0, hb0, hb1,
      /*yout_bf=*/y0_bf, /*yout_f32=*/nullptr,
      h_n, c_n, counters + 0);

  lstm_layer_kernel<<<NBLK, NTHR, shmem, stream>>>(
      y0_bf, W_ih1, W_hh1, b1, hb0, hb1,
      /*yout_bf=*/nullptr, /*yout_f32=*/y1,
      h_n + hc_elems, c_n + hc_elems, counters + 1);
}
